// AdditiveAttention_16338055594009
// MI455X (gfx1250) — compile-verified, matrix operand order checked by reading, not run on hardware
//
#include <hip/hip_runtime.h>
#include <hip/hip_bf16.h>

typedef __attribute__((ext_vector_type(16))) __bf16 v16bf;
typedef __attribute__((ext_vector_type(8)))  __bf16 v8bf;
typedef __attribute__((ext_vector_type(8)))  float  v8f;
typedef __attribute__((ext_vector_type(4)))  float  v4f;

#define B_   4
#define Q_   512
#define KV_  512
#define QS_  512
#define H_   256
#define VD_  512

#define LOG2E 1.4426950408889634f

// tanh(x) = 1 - 2/(exp(2x)+1); v_exp_f32 + v_rcp_f32, saturates correctly.
__device__ __forceinline__ float fast_tanh(float x) {
    float e = __builtin_amdgcn_exp2f(x * (2.0f * LOG2E));
    float r = __builtin_amdgcn_rcpf(e + 1.0f);
    return 1.0f - 2.0f * r;
}

// ---- CDNA5 async copy global -> LDS (ASYNCcnt-tracked, no VGPR data) ------
__device__ __forceinline__ void async_ld_b128(unsigned lds_off, const void* gptr) {
    asm volatile("global_load_async_to_lds_b128 %0, %1, off"
                 :: "v"(lds_off), "v"(gptr) : "memory");
}
// ASYNC loads complete in order: waiting <=8 with 8 newly-issued outstanding
// guarantees the previous stage's 8 loads have landed in LDS.
__device__ __forceinline__ void wait_async8() {
    asm volatile("s_wait_asynccnt 0x8" ::: "memory");
}
__device__ __forceinline__ void wait_async0() {
    asm volatile("s_wait_asynccnt 0x0" ::: "memory");
}
// Low 32 bits of a generic pointer into LDS == LDS byte offset (aperture layout).
__device__ __forceinline__ unsigned lds_off(const void* p) {
    return (unsigned)(unsigned long long)p;
}

// ---------------------------------------------------------------------------
// Prep kernel A: fp32 row-major -> bf16 row-major (8 elems / thread).
__global__ __launch_bounds__(256) void cvt_bf16_kernel(
    const float* __restrict__ src, __bf16* __restrict__ dst, int n8)
{
    int i = blockIdx.x * 256 + threadIdx.x;
    if (i >= n8) return;
    const v4f* s = (const v4f*)src + (size_t)i * 2;
    v4f a = s[0], b = s[1];
    v8bf o;
    o[0] = (__bf16)a[0]; o[1] = (__bf16)a[1]; o[2] = (__bf16)a[2]; o[3] = (__bf16)a[3];
    o[4] = (__bf16)b[0]; o[5] = (__bf16)b[1]; o[6] = (__bf16)b[2]; o[7] = (__bf16)b[3];
    ((v8bf*)dst)[i] = o;
}

// Prep kernel B: pack a row-major KxN fp32 matrix into WMMA-B fragment-major
// bf16: dst[((ntile*nk + kchunk)*32 + lane) * 16 + elem]. Strided gather
// happens HERE, once, not in the GEMM hot loops.
__global__ __launch_bounds__(256) void pack_bfrag_kernel(
    const float* __restrict__ Bm, __bf16* __restrict__ dst, int K, int N)
{
    const size_t bat = (size_t)blockIdx.y * K * N;
    int t    = blockIdx.x * 256 + threadIdx.x;
    int lane = t & 31;
    int tile = t >> 5;
    int nk   = K >> 5;
    int ntile  = tile / nk;
    int kchunk = tile - ntile * nk;
    if (ntile >= (N >> 4)) return;
    int n   = lane & 15;
    int klo = (lane < 16) ? 0 : 8;
    const float* p = Bm + bat + (size_t)(kchunk * 32 + klo) * N + ntile * 16 + n;
    v16bf f;
#pragma unroll
    for (int e = 0; e < 8; ++e) {
        f[e]     = (__bf16)p[(size_t)e * N];          // K = klo + e
        f[8 + e] = (__bf16)p[(size_t)(16 + e) * N];   // K = 16 + klo + e
    }
    ((v16bf*)(dst + bat))[(size_t)tile * 32 + lane] = f;
}

// ---------------------------------------------------------------------------
// A fragment (16x32) from bf16 row-major: two aligned 16B loads per lane.
__device__ __forceinline__ v16bf load_a_frag_bf(const __bf16* __restrict__ pa) {
    v8bf lo = *(const v8bf*)(pa);
    v8bf hi = *(const v8bf*)(pa + 16);
    return __builtin_shufflevector(lo, hi, 0,1,2,3,4,5,6,7,8,9,10,11,12,13,14,15);
}

// Projection GEMM: OUT(MxN) f32 = Xbf(MxK) * Wpacked(KxN).
// grid.x = M/16; 8 waves, 2 N-tiles each (N = 256).
__global__ __launch_bounds__(256) void proj_gemm_kernel(
    const __bf16* __restrict__ Xbf, const __bf16* __restrict__ Wp,
    float* __restrict__ OUT, int M, int K, int N)
{
    const int lane = threadIdx.x & 31;
    const int wave = threadIdx.x >> 5;
    const int nk   = K >> 5;
    const int m    = lane & 15;
    const int klo  = (lane < 16) ? 0 : 8;
    const __bf16* pa = Xbf + ((size_t)blockIdx.x * 16 + m) * K + klo;
    const v16bf* Wv0 = (const v16bf*)Wp + ((size_t)(wave * 2)     * nk) * 32 + lane;
    const v16bf* Wv1 = (const v16bf*)Wp + ((size_t)(wave * 2 + 1) * nk) * 32 + lane;

    v8f c0 = {}; v8f c1 = {};
    for (int kc = 0; kc < nk; ++kc) {
        v16bf a  = load_a_frag_bf(pa);  pa += 32;
        v16bf b0 = Wv0[kc * 32];
        v16bf b1 = Wv1[kc * 32];
        c0 = __builtin_amdgcn_wmma_f32_16x16x32_bf16(false, a, false, b0,
                                                     (short)0, c0, false, false);
        c1 = __builtin_amdgcn_wmma_f32_16x16x32_bf16(false, a, false, b1,
                                                     (short)0, c1, false, false);
    }
    int n  = lane & 15;
    int mb = (lane < 16) ? 0 : 8;
    int col0 = wave * 32, col1 = col0 + 16;
    float* Crow = OUT + (size_t)(blockIdx.x * 16 + mb) * N;
#pragma unroll
    for (int r = 0; r < 8; ++r) {
        Crow[(size_t)r * N + col0 + n] = c0[r];
        Crow[(size_t)r * N + col1 + n] = c1[r];
    }
}

// ---------------------------------------------------------------------------
// Fused: tanh-scores + masked softmax + attn @ values. One block per
// (b, 16-query tile); 256 threads = 8 waves. kf staging is double-buffered
// through ASYNCcnt so DMA streams behind the TRANS-bound tanh loop.
__global__ __launch_bounds__(256) void attn_fused_kernel(
    const float* __restrict__ qf, const float* __restrict__ kf,
    const float* __restrict__ wv, const __bf16* __restrict__ Vp,
    const int*   __restrict__ valid_lens, float* __restrict__ out)
{
    __shared__ float s_qf[16][260];       // stride 260: 16B-aligned rows, banks spread
    __shared__ float s_kf[2][32][260];    // double buffer
    __shared__ float s_wv[H_];
    __shared__ float s_attn[16][516];

    const int tid  = threadIdx.x;
    const int lane = tid & 31;
    const int wave = tid >> 5;
    const int b     = blockIdx.y;
    const int qbase = blockIdx.x * 16;
    const int vl    = valid_lens[b];
    const float* kfb = kf + (size_t)b * KV_ * H_;
    const v16bf* Vb  = (const v16bf*)Vp + (size_t)b * (32 * 16) * 32;

    s_wv[tid] = wv[tid];
    // async-stage qf tile (16 x 256 f32): 4 x b128 per thread
    {
        const float* src = qf + (size_t)(b * Q_ + qbase) * H_;
#pragma unroll
        for (int i = tid; i < 16 * (H_ / 4); i += 256) {
            int r = i >> 6, c = (i & 63) * 4;
            async_ld_b128(lds_off(&s_qf[r][c]), src + (size_t)r * H_ + c);
        }
    }
    // async-stage kf chunk 0 into buffer 0: 8 x b128 per thread
#pragma unroll
    for (int i = tid; i < 32 * (H_ / 4); i += 256) {
        int r = i >> 6, c = (i & 63) * 4;
        async_ld_b128(lds_off(&s_kf[0][r][c]), kfb + (size_t)r * H_ + c);
    }
    // warm L2/WGP$ with this batch's packed values (512 KB) while scores run
    {
        const char* pv = (const char*)Vb;
        for (int i = tid; i < 4096; i += 256)
            __builtin_prefetch(pv + (size_t)i * 128, 0, 1);
    }

    // ---- scores: 16q x 512k, reduce over h = 256 with tanh ----
    const int q   = tid & 15;
    const int kl0 = tid >> 4;
    for (int kc = 0; kc < 16; ++kc) {
        const int buf = kc & 1;
        if (kc < 15) {
            // issue next chunk into the other buffer (last read 2 barriers ago)
#pragma unroll
            for (int i = tid; i < 32 * (H_ / 4); i += 256) {
                int r = i >> 6, c = (i & 63) * 4;
                async_ld_b128(lds_off(&s_kf[buf ^ 1][r][c]),
                              kfb + (size_t)((kc + 1) * 32 + r) * H_ + c);
            }
            wait_async8();   // current chunk (and qf on kc==0) has landed
        } else {
            wait_async0();
        }
        __syncthreads();
        float acc0 = 0.f, acc1 = 0.f;
#pragma unroll 4
        for (int h = 0; h < H_; ++h) {
            float w  = s_wv[h];
            float qv = s_qf[q][h];
            acc0 += w * fast_tanh(qv + s_kf[buf][kl0][h]);
            acc1 += w * fast_tanh(qv + s_kf[buf][kl0 + 16][h]);
        }
        s_attn[q][kc * 32 + kl0]      = acc0;
        s_attn[q][kc * 32 + kl0 + 16] = acc1;
        __syncthreads();     // all reads of s_kf[buf] done before it is re-staged
    }

    // ---- masked softmax over KV, wave32 shuffle reductions ----
#pragma unroll
    for (int rr = 0; rr < 2; ++rr) {
        int row = wave * 2 + rr;
        float vals[16];
        float mx = -3.4e38f;
#pragma unroll
        for (int i = 0; i < 16; ++i) {
            int k = lane + 32 * i;
            float s = s_attn[row][k];
            s = (k < vl) ? s : -1e6f;
            vals[i] = s;
            mx = fmaxf(mx, s);
        }
#pragma unroll
        for (int off = 16; off > 0; off >>= 1)
            mx = fmaxf(mx, __shfl_xor(mx, off, 32));
        float sum = 0.f;
#pragma unroll
        for (int i = 0; i < 16; ++i) {
            float e = __builtin_amdgcn_exp2f((vals[i] - mx) * LOG2E);
            vals[i] = e;
            sum += e;
        }
#pragma unroll
        for (int off = 16; off > 0; off >>= 1)
            sum += __shfl_xor(sum, off, 32);
        float inv = __builtin_amdgcn_rcpf(sum);
#pragma unroll
        for (int i = 0; i < 16; ++i)
            s_attn[row][lane + 32 * i] = vals[i] * inv;
    }
    __syncthreads();

    // ---- out tile (16x512) = attn (16x512) @ values (512x512), bf16 WMMA ----
    float* outb = out + (size_t)(b * Q_ + qbase) * VD_;
    v8f acc[4] = {v8f{}, v8f{}, v8f{}, v8f{}};
    const int m   = lane & 15;
    const int klo = (lane < 16) ? 0 : 8;
    for (int kc = 0; kc < 16; ++kc) {
        const float* pa = &s_attn[m][kc * 32 + klo];
        v16bf a;
#pragma unroll
        for (int j = 0; j < 4; ++j) {
            a[2*j]       = (__bf16)pa[2*j];
            a[2*j+1]     = (__bf16)pa[2*j + 1];
            a[8 + 2*j]   = (__bf16)pa[16 + 2*j];
            a[8 + 2*j+1] = (__bf16)pa[16 + 2*j + 1];
        }
#pragma unroll
        for (int t = 0; t < 4; ++t) {
            int ntile = wave * 4 + t;
            v16bf bf = Vb[((size_t)ntile * 16 + kc) * 32 + lane];
            acc[t] = __builtin_amdgcn_wmma_f32_16x16x32_bf16(false, a, false, bf,
                                                             (short)0, acc[t],
                                                             false, false);
        }
    }
    int n  = lane & 15;
    int mb = (lane < 16) ? 0 : 8;
#pragma unroll
    for (int t = 0; t < 4; ++t) {
        int col = (wave * 4 + t) * 16;
#pragma unroll
        for (int r = 0; r < 8; ++r)
            outb[(size_t)(mb + r) * VD_ + col + n] = acc[t][r];
    }
}

// ---------------------------------------------------------------------------
extern "C" void kernel_launch(void* const* d_in, const int* in_sizes, int n_in,
                              void* d_out, int out_size, void* d_ws, size_t ws_size,
                              hipStream_t stream) {
    const float* queries = (const float*)d_in[0];   // (B, Q, QS)
    const float* keys    = (const float*)d_in[1];   // (B, KV, KS)
    const float* values  = (const float*)d_in[2];   // (B, KV, VD)
    const int*   vlens   = (const int*)  d_in[3];   // (B,)
    const float* W_q     = (const float*)d_in[4];   // (QS, H)
    const float* W_k     = (const float*)d_in[5];   // (KS, H)
    const float* w_v     = (const float*)d_in[6];   // (H,)
    float* out = (float*)d_out;

    char* w = (char*)d_ws;
    float*  qf  = (float*)w;   w += (size_t)B_ * Q_  * H_  * 4;   // 2 MB
    float*  kfp = (float*)w;   w += (size_t)B_ * KV_ * H_  * 4;   // 2 MB
    __bf16* qbf = (__bf16*)w;  w += (size_t)B_ * Q_  * QS_ * 2;   // 2 MB
    __bf16* kbf = (__bf16*)w;  w += (size_t)B_ * KV_ * QS_ * 2;   // 2 MB
    __bf16* pWq = (__bf16*)w;  w += (size_t)QS_ * H_ * 2;         // 256 KB
    __bf16* pWk = (__bf16*)w;  w += (size_t)QS_ * H_ * 2;         // 256 KB
    __bf16* pV  = (__bf16*)w;  w += (size_t)B_ * KV_ * VD_ * 2;   // 2 MB

    // Prep: bf16 conversions + fragment packing (all tiny, L2-resident).
    const int nq8 = (B_ * Q_ * QS_) / 8;
    cvt_bf16_kernel<<<dim3(nq8 / 256), 256, 0, stream>>>(queries, qbf, nq8);
    cvt_bf16_kernel<<<dim3(nq8 / 256), 256, 0, stream>>>(keys,    kbf, nq8);
    pack_bfrag_kernel<<<dim3(32, 1), 256, 0, stream>>>(W_q, pWq, QS_, H_);
    pack_bfrag_kernel<<<dim3(32, 1), 256, 0, stream>>>(W_k, pWk, QS_, H_);
    pack_bfrag_kernel<<<dim3(64, B_), 256, 0, stream>>>(values, pV, KV_, VD_);

    // Projections (WMMA bf16), then fused scores/softmax/output.
    proj_gemm_kernel<<<dim3((B_ * Q_) / 16), 256, 0, stream>>>(
        qbf, pWq, qf, B_ * Q_, QS_, H_);
    proj_gemm_kernel<<<dim3((B_ * KV_) / 16), 256, 0, stream>>>(
        kbf, pWk, kfp, B_ * KV_, QS_, H_);
    attn_fused_kernel<<<dim3(Q_ / 16, B_), 256, 0, stream>>>(
        qf, kfp, w_v, pV, vlens, out);
}
